// RSSM_17849884082580
// MI455X (gfx1250) — matrix operand order repaired and checked by reading, not compiled, not run on hardware
//
#include <hip/hip_runtime.h>

typedef __attribute__((ext_vector_type(16))) _Float16 v16h;
typedef __attribute__((ext_vector_type(8)))  float    v8f;

#define DEVINL __device__ __forceinline__

DEVINL float eluf(float x)     { return x > 0.f ? x : (__expf(x) - 1.f); }
DEVINL float sigmoidf(float x) { return 1.f / (1.f + __expf(-x)); }

// ---------------------------------------------------------------------------
// Async global->LDS 16B copy (GLOBAL_LOAD_ASYNC_TO_LDS_B128, GV mode).
// ISA 10.2: for generic pointers in the LDS aperture the low 32 bits are the
// LDS byte address, which is what the instruction's VDST operand carries.
// Tracked on ASYNCcnt; s_wait_asynccnt 0 before the barrier makes it visible.
// ---------------------------------------------------------------------------
DEVINL void async_copy16(void* lds_dst, const void* gsrc) {
  unsigned lds = (unsigned)(size_t)lds_dst;
  asm volatile("global_load_async_to_lds_b128 %0, %1, off"
               :: "v"(lds), "v"((unsigned long long)(size_t)gsrc)
               : "memory");
}
DEVINL void async_wait0() {
  asm volatile("s_wait_asynccnt 0x0" ::: "memory");
}

// ---------------------------------------------------------------------------
// One 16x16 output tile of  D = X(MxK) * W(NxK)^T  accumulated in f32 via
// v_wmma_f32_16x16x32_f16.  X row-major [M][lda], W row-major [N][ldw]
// (torch Linear weight layout), K multiple of 32.
// A frag (16x32 f16, ISA 7.12.2): lane m=lane&15, VGPR v: K = 16*(v>>2) + 8*(lane>>4) + 2*(v&3)
// B frag (32x16 f16):             lane n=lane&15, VGPR v: K = 16*(v>>2) + 8*(lane>>4) + 2*(v&3)
// ---------------------------------------------------------------------------
DEVINL v8f wmma_tile(const _Float16* A, int lda, const _Float16* W, int ldw,
                     int m0, int n0, int K, v8f acc) {
  const int lane = threadIdx.x & 31;
  const int hl   = lane >> 4;
  const _Float16* ar = A + (m0 + (lane & 15)) * lda;
  const _Float16* br = W + (n0 + (lane & 15)) * ldw;
  for (int k0 = 0; k0 < K; k0 += 32) {
    v16h a, b;
#pragma unroll
    for (int v = 0; v < 8; ++v) {
      const int ka = k0 + ((v >> 2) << 4) + (hl << 3) + ((v & 3) << 1);
      const int kb = k0 + ((v >> 2) << 4) + (hl << 3) + ((v & 3) << 1);
      a[2 * v]     = ar[ka];
      a[2 * v + 1] = ar[ka + 1];
      b[2 * v]     = br[kb];
      b[2 * v + 1] = br[kb + 1];
    }
    acc = __builtin_amdgcn_wmma_f32_16x16x32_f16(false, a, false, b,
                                                 (short)0, acc, false, false);
  }
  return acc;
}

// C/D f32 layout: VGPR r -> row m0 + r + 8*(lane>>4), col n0 + (lane&15)
DEVINL void store_f16_elu(_Float16* D, int ldd, const float* bias,
                          int m0, int n0, v8f acc) {
  const int lane = threadIdx.x & 31, hl = lane >> 4;
  const int col = n0 + (lane & 15);
  const float bv = bias[col];
#pragma unroll
  for (int r = 0; r < 8; ++r)
    D[(m0 + r + 8 * hl) * ldd + col] = (_Float16)eluf(acc[r] + bv);
}

DEVINL void store_f32_bias(float* D, int ldd, const float* bias,
                           int m0, int n0, v8f acc) {
  const int lane = threadIdx.x & 31, hl = lane >> 4;
  const int col = n0 + (lane & 15);
  const float bv = bias[col];
#pragma unroll
  for (int r = 0; r < 8; ++r)
    D[(m0 + r + 8 * hl) * ldd + col] = acc[r] + bv;
}

// transposed store for conv GEMMs: tile row = spatial pos, col = out channel;
// dest layout [oc][pos] so it is directly the next layer's im2col source.
DEVINL void store_tr_relu_f16(_Float16* D, int ldd, const float* bias,
                              int m0, int n0, v8f acc) {
  const int lane = threadIdx.x & 31, hl = lane >> 4;
  const int col = n0 + (lane & 15);
  const float bv = bias[col];
#pragma unroll
  for (int r = 0; r < 8; ++r) {
    int row = m0 + r + 8 * hl;
    D[col * ldd + row] = (_Float16)fmaxf(acc[r] + bv, 0.f);
  }
}

// 3-layer MLP: [32 x ldx] -> ELU 128 -> ELU 128 -> 64 raw (into OutP f32)
DEVINL void mlp3(const _Float16* X, int ldx,
                 const _Float16* W1, const float* B1,
                 const _Float16* W2, const float* B2,
                 const _Float16* W3, const float* B3,
                 _Float16* H1, _Float16* H2, float* OutP, int wave) {
  for (int job = wave; job < 16; job += 8) {
    int m0 = (job >> 3) * 16, n0 = (job & 7) * 16;
    v8f acc = {};
    acc = wmma_tile(X, ldx, W1, ldx, m0, n0, ldx, acc);
    store_f16_elu(H1, 128, B1, m0, n0, acc);
  }
  __syncthreads();
  for (int job = wave; job < 16; job += 8) {
    int m0 = (job >> 3) * 16, n0 = (job & 7) * 16;
    v8f acc = {};
    acc = wmma_tile(H1, 128, W2, 128, m0, n0, 128, acc);
    store_f16_elu(H2, 128, B2, m0, n0, acc);
  }
  __syncthreads();
  for (int job = wave; job < 8; job += 8) {
    int m0 = (job >> 2) * 16, n0 = (job & 3) * 16;
    v8f acc = {};
    acc = wmma_tile(H2, 128, W3, 128, m0, n0, 128, acc);
    store_f32_bias(OutP, 64, B3, m0, n0, acc);
  }
  __syncthreads();
}

// ---------------------------------------------------------------------------
// Weight convert + pad (f32 [rows][sc] -> f16 [rows][dc], zero-padded)
// ---------------------------------------------------------------------------
__global__ __launch_bounds__(256) void k_cvt(const float* __restrict__ src,
                                             _Float16* __restrict__ dst,
                                             int rows, int sc, int dc) {
  int i = blockIdx.x * 256 + threadIdx.x;
  if (i < rows * dc) {
    int r = i / dc, c = i - r * dc;
    dst[i] = (c < sc) ? (_Float16)src[r * sc + c] : (_Float16)0.f;
  }
}

// ConvTranspose2d weight repack: src[ic][oc][4][4] -> dst[oc][ic*16 + j] (f16)
__global__ __launch_bounds__(256) void k_cvt_dw(const float* __restrict__ src,
                                                _Float16* __restrict__ dst,
                                                int IC, int OC) {
  int i = blockIdx.x * 256 + threadIdx.x;
  if (i < IC * OC * 16) {
    int j = i & 15, rest = i >> 4;
    int oc = rest / IC, ic = rest - oc * IC;
    dst[i] = (_Float16)src[(ic * OC + oc) * 16 + j];
  }
}

// ---------------------------------------------------------------------------
// Encoder: one block per frame. conv1 VALU; conv2/conv3 as im2col WMMA GEMMs.
// conv(k4,s2,p1): 1x32x32 -> 32x16x16 -> 64x8x8 -> 64x4x4, relu each; f16 out
// ---------------------------------------------------------------------------
__global__ __launch_bounds__(256) void k_enc(
    const float* __restrict__ obs,
    const float* __restrict__ w1, const float* __restrict__ b1,
    const _Float16* __restrict__ w2c, const float* __restrict__ b2,
    const _Float16* __restrict__ w3c, const float* __restrict__ b3,
    _Float16* __restrict__ feat) {
  const int f = blockIdx.x, tid = threadIdx.x, wave = tid >> 5;
  __shared__ float    x0[1024];     // 1x32x32          (4 KB)
  __shared__ _Float16 x1[8192];     // [32][256] pos    (16 KB)
  __shared__ _Float16 x2[4096];     // [64][64] pos     (8 KB)
  __shared__ _Float16 col[8192];    // im2col chunk     (16 KB)

  for (int i = tid; i < 1024; i += 256) x0[i] = obs[f * 1024 + i];
  __syncthreads();

  // conv1 (K=16, VALU): 32x16x16
  for (int idx = tid; idx < 8192; idx += 256) {
    int oc = idx >> 8, oh = (idx >> 4) & 15, ow = idx & 15;
    float s = b1[oc];
    for (int kh = 0; kh < 4; ++kh) {
      int ih = oh * 2 - 1 + kh; if ((unsigned)ih >= 32u) continue;
      for (int kw = 0; kw < 4; ++kw) {
        int iw = ow * 2 - 1 + kw; if ((unsigned)iw >= 32u) continue;
        s += x0[ih * 32 + iw] * w1[oc * 16 + kh * 4 + kw];
      }
    }
    x1[idx] = (_Float16)fmaxf(s, 0.f);
  }
  __syncthreads();

  // conv2 GEMM: M=64 pos, N=64 oc, K=512 in 4 chunks of 128 (8 ic/chunk)
  {
    v8f acc[2] = {{}, {}};
    for (int c = 0; c < 4; ++c) {
      for (int i = tid; i < 8192; i += 256) {
        int p = i >> 7, kk = i & 127;
        int ic = c * 8 + (kk >> 4), kh = (kk >> 2) & 3, kw = kk & 3;
        int oh = p >> 3, ow = p & 7;
        int ih = oh * 2 - 1 + kh, iw = ow * 2 - 1 + kw;
        col[i] = ((unsigned)ih < 16u && (unsigned)iw < 16u)
                     ? x1[ic * 256 + ih * 16 + iw] : (_Float16)0.f;
      }
      __syncthreads();
#pragma unroll
      for (int j = 0; j < 2; ++j) {
        int job = wave + j * 8;                 // 16 jobs: mt(4) x nt(4)
        int m0 = (job >> 2) * 16, n0 = (job & 3) * 16;
        acc[j] = wmma_tile(col, 128, w2c + c * 128, 512, m0, n0, 128, acc[j]);
      }
      __syncthreads();
    }
#pragma unroll
    for (int j = 0; j < 2; ++j) {
      int job = wave + j * 8;
      store_tr_relu_f16(x2, 64, b2, (job >> 2) * 16, (job & 3) * 16, acc[j]);
    }
  }
  __syncthreads();

  // conv3 GEMM: M=16 pos, N=64 oc, K=1024 in 8 chunks of 128 (8 ic/chunk)
  {
    v8f acc = {};
    for (int c = 0; c < 8; ++c) {
      for (int i = tid; i < 2048; i += 256) {
        int p = i >> 7, kk = i & 127;
        int ic = c * 8 + (kk >> 4), kh = (kk >> 2) & 3, kw = kk & 3;
        int oh = p >> 2, ow = p & 3;
        int ih = oh * 2 - 1 + kh, iw = ow * 2 - 1 + kw;
        col[i] = ((unsigned)ih < 8u && (unsigned)iw < 8u)
                     ? x2[ic * 64 + ih * 8 + iw] : (_Float16)0.f;
      }
      __syncthreads();
      if (wave < 4)                              // 4 jobs: nt(4)
        acc = wmma_tile(col, 128, w3c + c * 128, 1024, 0, wave * 16, 128, acc);
      __syncthreads();
    }
    if (wave < 4)                                // feat[f][oc*16 + pos]
      store_tr_relu_f16(feat + f * 1024, 16, b3, 0, wave * 16, acc);
  }
}

// ---------------------------------------------------------------------------
// feat(2048x1024 f16) @ encPw(64x1024 f16)^T + b -> embed(2048x64 f32)
// LDS-staged: block owns a 32-row A slab, staged in 4 K-chunks of 256 via
// async global->LDS copies; 8 waves each own one 16x16 output tile.
// ---------------------------------------------------------------------------
__global__ __launch_bounds__(256) void k_gemm_embed(
    const _Float16* __restrict__ feat, const _Float16* __restrict__ w,
    const float* __restrict__ bias, float* __restrict__ embed) {
  const int tid = threadIdx.x, wave = tid >> 5;
  const int r0 = blockIdx.x * 32;
  __shared__ alignas(16) _Float16 As[32 * 256];   // 16 KB chunk
  const int mt = wave >> 2, nt = wave & 3;        // 8 jobs: mt(2) x nt(4)
  v8f acc = {};
  for (int c = 0; c < 4; ++c) {
    for (int i = tid; i < 512; i += 256) {        // 32 rows x 16 segs of 16B
      int row = i >> 4, seg = i & 15;
      async_copy16(&As[row * 256 + seg * 8],
                   feat + (size_t)(r0 + row) * 1024 + c * 256 + seg * 8);
    }
    async_wait0();
    __syncthreads();
    acc = wmma_tile(As, 256, w + c * 256, 1024, mt * 16, nt * 16, 256, acc);
    __syncthreads();
  }
  store_f32_bias(embed + r0 * 64, 64, bias, mt * 16, nt * 16, acc);
}

// ---------------------------------------------------------------------------
// Sequential RSSM: ONE persistent workgroup (8 waves), state in LDS.
// ---------------------------------------------------------------------------
__global__ __launch_bounds__(256) void k_rssm(
    const float* __restrict__ actions, const float* __restrict__ noise,
    const float* __restrict__ embed,
    const _Float16* posW1, const _Float16* posW2, const _Float16* posW3,
    const float* posB1, const float* posB2, const float* posB3,
    const _Float16* priW1, const _Float16* priW2, const _Float16* priW3,
    const float* priB1, const float* priB2, const float* priB3,
    const _Float16* wih, const _Float16* whh,
    const float* bih, const float* bhh,
    _Float16* __restrict__ hz16,
    float* o_zs, float* o_prm, float* o_prls, float* o_pm, float* o_pls,
    float* o_hf) {
  const int tid = threadIdx.x, wave = tid >> 5;
  __shared__ float    sh_h[4096];   // h  (32x128 f32), persistent
  __shared__ _Float16 sh_z[1024];   // z  (32x32 f16),  persistent per step
  __shared__ alignas(16) char scr[45056];  // phase-aliased scratch
  _Float16* Xpos = (_Float16*)scr;               // A: [32][192]
  _Float16* Xpri = (_Float16*)scr;               // B: [32][160]
  _Float16* H1   = (_Float16*)(scr + 12288);     // A/B: [32][128] f16
  _Float16* H2   = (_Float16*)(scr + 20480);     // A/B: [32][128] f16
  float*    OutP = (float*)(scr + 28672);        // A/B: [32][64]  f32
  _Float16* Xh   = (_Float16*)scr;               // C: h f16 [32][128]
  _Float16* Xx   = (_Float16*)(scr + 8192);      // C: gru x [32][64]
  float*    rbuf = (float*)(scr + 12288);        // C: r gate [32][128]
  float*    ubuf = (float*)(scr + 28672);        // C: u gate [32][128]

  // warm near caches for recurrent weights (global_prefetch_b8); they are
  // re-read 64x by this single persistent workgroup.
  for (int i = tid * 64; i < 128 * 192; i += 256 * 64) __builtin_prefetch(posW1 + i, 0, 3);
  for (int i = tid * 64; i < 128 * 128; i += 256 * 64) __builtin_prefetch(posW2 + i, 0, 3);
  for (int i = tid * 64; i < 64 * 128;  i += 256 * 64) __builtin_prefetch(posW3 + i, 0, 3);
  for (int i = tid * 64; i < 128 * 160; i += 256 * 64) __builtin_prefetch(priW1 + i, 0, 3);
  for (int i = tid * 64; i < 128 * 128; i += 256 * 64) __builtin_prefetch(priW2 + i, 0, 3);
  for (int i = tid * 64; i < 64 * 128;  i += 256 * 64) __builtin_prefetch(priW3 + i, 0, 3);
  for (int i = tid * 64; i < 384 * 64;  i += 256 * 64) __builtin_prefetch(wih + i, 0, 3);
  for (int i = tid * 64; i < 384 * 128; i += 256 * 64) __builtin_prefetch(whh + i, 0, 3);

  for (int i = tid; i < 4096; i += 256) sh_h[i] = 0.f;
  __syncthreads();

  for (int t = 0; t < 64; ++t) {
    // pre-update h -> hz buffer; build posterior input [h | e_t]
    for (int i = tid; i < 4096; i += 256) {
      int b = i >> 7, c = i & 127;
      hz16[(t * 32 + b) * 160 + c] = (_Float16)sh_h[i];
    }
    for (int i = tid; i < 32 * 192; i += 256) {
      int b = i / 192, c = i - b * 192;
      float v = (c < 128) ? sh_h[b * 128 + c]
                          : embed[(b * 64 + t) * 64 + (c - 128)];
      Xpos[i] = (_Float16)v;
    }
    __syncthreads();
    mlp3(Xpos, 192, posW1, posB1, posW2, posB2, posW3, posB3, H1, H2, OutP, wave);
    // posterior sample: z = pm + exp(clip(pls))*eps
    for (int i = tid; i < 1024; i += 256) {
      int b = i >> 5, s = i & 31;
      float pm  = OutP[b * 64 + s];
      float pls = fminf(fmaxf(OutP[b * 64 + 32 + s], -5.f), 2.f);
      float z   = pm + __expf(pls) * noise[(t * 32 + b) * 32 + s];
      int oi = (b * 64 + t) * 32 + s;
      o_pm[oi] = pm; o_pls[oi] = pls; o_zs[oi] = z;
      sh_z[i] = (_Float16)z;
      hz16[(t * 32 + b) * 160 + 128 + s] = (_Float16)z;
    }
    __syncthreads();
    // prior input [h | a | 0-pad]
    for (int i = tid; i < 32 * 160; i += 256) {
      int b = i / 160, c = i - b * 160;
      float v = (c < 128) ? sh_h[b * 128 + c]
              : (c < 130) ? actions[(b * 64 + t) * 2 + (c - 128)] : 0.f;
      Xpri[i] = (_Float16)v;
    }
    __syncthreads();
    mlp3(Xpri, 160, priW1, priB1, priW2, priB2, priW3, priB3, H1, H2, OutP, wave);
    for (int i = tid; i < 1024; i += 256) {
      int b = i >> 5, s = i & 31;
      int oi = (b * 64 + t) * 32 + s;
      o_prm[oi]  = OutP[b * 64 + s];
      o_prls[oi] = fminf(fmaxf(OutP[b * 64 + 32 + s], -5.f), 2.f);
    }
    // GRU inputs: stale-f16 h copy + [z | a | 0-pad]
    for (int i = tid; i < 4096; i += 256) Xh[i] = (_Float16)sh_h[i];
    for (int i = tid; i < 2048; i += 256) {
      int b = i >> 6, c = i & 63;
      float v = (c < 32) ? (float)sh_z[b * 32 + c]
              : (c < 34) ? actions[(b * 64 + t) * 2 + (c - 32)] : 0.f;
      Xx[i] = (_Float16)v;
    }
    __syncthreads();
    // GRU pass 1: r and u gates (gate cols 0..255)
    for (int job = wave; job < 32; job += 8) {
      int m0 = (job & 1) * 16, n0 = (job >> 1) * 16;
      v8f gi = {}; gi = wmma_tile(Xx, 64, wih, 64, m0, n0, 64, gi);
      v8f gh = {}; gh = wmma_tile(Xh, 128, whh, 128, m0, n0, 128, gh);
      const int lane = tid & 31, hl = lane >> 4;
      const int col = n0 + (lane & 15);
      const float bi = bih[col], bh = bhh[col];
#pragma unroll
      for (int r = 0; r < 8; ++r) {
        int row = m0 + r + 8 * hl;
        float g = sigmoidf(gi[r] + bi + gh[r] + bh);
        if (col < 128) rbuf[row * 128 + col] = g;
        else           ubuf[row * 128 + col - 128] = g;
      }
    }
    __syncthreads();
    // GRU pass 2: n gate + state update (cols 256..383); reads stale Xh
    for (int job = wave; job < 16; job += 8) {
      int m0 = (job & 1) * 16, n0 = 256 + (job >> 1) * 16;
      v8f gi = {}; gi = wmma_tile(Xx, 64, wih, 64, m0, n0, 64, gi);
      v8f gh = {}; gh = wmma_tile(Xh, 128, whh, 128, m0, n0, 128, gh);
      const int lane = tid & 31, hl = lane >> 4;
      const int col = n0 + (lane & 15), j = col - 256;
      const float bi = bih[col], bh = bhh[col];
#pragma unroll
      for (int r = 0; r < 8; ++r) {
        int row = m0 + r + 8 * hl;
        float rv = rbuf[row * 128 + j];
        float u  = ubuf[row * 128 + j];
        float n  = tanhf(gi[r] + bi + rv * (gh[r] + bh));
        float ho = sh_h[row * 128 + j];
        sh_h[row * 128 + j] = (1.f - u) * n + u * ho;
      }
    }
    __syncthreads();
  }
  for (int i = tid; i < 4096; i += 256) o_hf[i] = sh_h[i];
}

// ---------------------------------------------------------------------------
// hz(2048x160 f16) @ decPw(1024x160 f16)^T + b -> declin(2048x1024 f32)
// LDS-staged: block owns one 16-row A slab (async-copied once, 5 KB),
// each of 8 waves computes 8 of the 64 n-tiles from LDS.  Cuts redundant
// global A reads 64x vs per-tile gathering.
// ---------------------------------------------------------------------------
__global__ __launch_bounds__(256) void k_gemm_dec(
    const _Float16* __restrict__ hz, const _Float16* __restrict__ w,
    const float* __restrict__ bias, float* __restrict__ y) {
  const int tid = threadIdx.x, wave = tid >> 5;
  const int r0 = blockIdx.x * 16;
  __shared__ alignas(16) _Float16 As[16 * 160];   // 5 KB
  if (tid < 160) {                                // 16 rows x 10 segs of 16B
    int row = tid / 10, seg = tid - row * 10;
    async_copy16(&As[row * 160 + seg * 8],
                 hz + (size_t)(r0 + row) * 160 + seg * 8);
  }
  async_wait0();
  __syncthreads();
  for (int nt = wave; nt < 64; nt += 8) {
    v8f acc = {};
    acc = wmma_tile(As, 160, w, 160, 0, nt * 16, 160, acc);
    store_f32_bias(y + r0 * 1024, 1024, bias, 0, nt * 16, acc);
  }
}

// ---------------------------------------------------------------------------
// Decoder: one block per (t,b) sample. deconv1/deconv2 as zero-masked
// im2col WMMA GEMMs; deconv3 (N=1) VALU + sigmoid.
// deconvT(k4,s2,p1): 64x4x4 -> 64x8x8 -> 32x16x16 -> 1x32x32
// ---------------------------------------------------------------------------
__global__ __launch_bounds__(256) void k_dec(
    const float* __restrict__ yin,
    const _Float16* __restrict__ w1p, const float* __restrict__ b1,
    const _Float16* __restrict__ w2p, const float* __restrict__ b2,
    const float* __restrict__ w3, const float* __restrict__ b3,
    float* __restrict__ recons) {
  const int smp = blockIdx.x, tid = threadIdx.x, wave = tid >> 5;
  __shared__ _Float16 y0[1024];     // [64][16]   (2 KB)
  __shared__ _Float16 y1[4096];     // [64][64]   (8 KB)
  __shared__ _Float16 y2[8192];     // [32][256]  (16 KB)
  __shared__ _Float16 col[16384];   // im2col     (32 KB)

  for (int i = tid; i < 1024; i += 256) y0[i] = (_Float16)yin[smp * 1024 + i];
  __syncthreads();

  // deconv1 GEMM: M=64 pos (8x8), N=64 oc, K=1024 in 8 chunks of 128
  {
    v8f acc[2] = {{}, {}};
    for (int c = 0; c < 8; ++c) {
      for (int i = tid; i < 8192; i += 256) {
        int p = i >> 7, kk = i & 127;
        int ic = c * 8 + (kk >> 4), kh = (kk >> 2) & 3, kw = kk & 3;
        int oh = p >> 3, ow = p & 7;
        int th = oh + 1 - kh, tw = ow + 1 - kw;
        bool ok = th >= 0 && !(th & 1) && (th >> 1) < 4 &&
                  tw >= 0 && !(tw & 1) && (tw >> 1) < 4;
        col[i] = ok ? y0[ic * 16 + (th >> 1) * 4 + (tw >> 1)] : (_Float16)0.f;
      }
      __syncthreads();
#pragma unroll
      for (int j = 0; j < 2; ++j) {
        int job = wave + j * 8;                 // 16 jobs: mt(4) x nt(4)
        int m0 = (job >> 2) * 16, n0 = (job & 3) * 16;
        acc[j] = wmma_tile(col, 128, w1p + c * 128, 1024, m0, n0, 128, acc[j]);
      }
      __syncthreads();
    }
#pragma unroll
    for (int j = 0; j < 2; ++j) {
      int job = wave + j * 8;
      store_tr_relu_f16(y1, 64, b1, (job >> 2) * 16, (job & 3) * 16, acc[j]);
    }
  }
  __syncthreads();

  // deconv2 GEMM: M=256 pos (16x16), N=32 oc, K=1024 in 16 chunks of 64
  {
    v8f acc[4] = {{}, {}, {}, {}};
    for (int c = 0; c < 16; ++c) {
      for (int i = tid; i < 16384; i += 256) {
        int p = i >> 6, kk = i & 63;
        int ic = c * 4 + (kk >> 4), kh = (kk >> 2) & 3, kw = kk & 3;
        int oh = p >> 4, ow = p & 15;
        int th = oh + 1 - kh, tw = ow + 1 - kw;
        bool ok = th >= 0 && !(th & 1) && (th >> 1) < 8 &&
                  tw >= 0 && !(tw & 1) && (tw >> 1) < 8;
        col[i] = ok ? y1[ic * 64 + (th >> 1) * 8 + (tw >> 1)] : (_Float16)0.f;
      }
      __syncthreads();
#pragma unroll
      for (int j = 0; j < 4; ++j) {
        int job = wave + j * 8;                 // 32 jobs: mt(16) x nt(2)
        int m0 = (job >> 1) * 16, n0 = (job & 1) * 16;
        acc[j] = wmma_tile(col, 64, w2p + c * 64, 1024, m0, n0, 64, acc[j]);
      }
      __syncthreads();
    }
#pragma unroll
    for (int j = 0; j < 4; ++j) {
      int job = wave + j * 8;
      store_tr_relu_f16(y2, 256, b2, (job >> 1) * 16, (job & 1) * 16, acc[j]);
    }
  }
  __syncthreads();

  // deconv3 (N=1) VALU + sigmoid; smp = t*32 + b
  const int t = smp >> 5, b = smp & 31;
  for (int idx = tid; idx < 1024; idx += 256) {
    int oh = idx >> 5, ow = idx & 31;
    float s = b3[0];
    for (int kh = 0; kh < 4; ++kh) {
      int th = oh + 1 - kh; if (th < 0 || (th & 1)) continue;
      int ih = th >> 1;     if (ih >= 16) continue;
      for (int kw = 0; kw < 4; ++kw) {
        int tw = ow + 1 - kw; if (tw < 0 || (tw & 1)) continue;
        int iw = tw >> 1;     if (iw >= 16) continue;
        for (int ic = 0; ic < 32; ++ic)
          s += (float)y2[ic * 256 + ih * 16 + iw] * w3[ic * 16 + kh * 4 + kw];
      }
    }
    recons[(b * 64 + t) * 1024 + idx] = sigmoidf(s);
  }
}

// ---------------------------------------------------------------------------
extern "C" void kernel_launch(void* const* d_in, const int* in_sizes, int n_in,
                              void* d_out, int out_size, void* d_ws, size_t ws_size,
                              hipStream_t stream) {
  (void)in_sizes; (void)n_in; (void)out_size; (void)ws_size;
  const float* obs   = (const float*)d_in[0];
  const float* acts  = (const float*)d_in[1];
  const float* noise = (const float*)d_in[2];
  const float* ew1 = (const float*)d_in[3];  const float* eb1 = (const float*)d_in[4];
  const float* ew2 = (const float*)d_in[5];  const float* eb2 = (const float*)d_in[6];
  const float* ew3 = (const float*)d_in[7];  const float* eb3 = (const float*)d_in[8];
  const float* encPw = (const float*)d_in[9];  const float* encPb = (const float*)d_in[10];
  const float* decPw = (const float*)d_in[11]; const float* decPb = (const float*)d_in[12];
  const float* dw1 = (const float*)d_in[13]; const float* db1 = (const float*)d_in[14];
  const float* dw2 = (const float*)d_in[15]; const float* db2 = (const float*)d_in[16];
  const float* dw3 = (const float*)d_in[17]; const float* db3 = (const float*)d_in[18];
  const float* gWih = (const float*)d_in[19]; const float* gWhh = (const float*)d_in[20];
  const float* gBih = (const float*)d_in[21]; const float* gBhh = (const float*)d_in[22];
  const float* pw1 = (const float*)d_in[23]; const float* pb1 = (const float*)d_in[24];
  const float* pw2 = (const float*)d_in[25]; const float* pb2 = (const float*)d_in[26];
  const float* pw3 = (const float*)d_in[27]; const float* pb3 = (const float*)d_in[28];
  const float* qw1 = (const float*)d_in[29]; const float* qb1 = (const float*)d_in[30];
  const float* qw2 = (const float*)d_in[31]; const float* qb2 = (const float*)d_in[32];
  const float* qw3 = (const float*)d_in[33]; const float* qb3 = (const float*)d_in[34];

  char* ws = (char*)d_ws;
  size_t off = 0;
  auto alloc = [&](size_t bytes) {
    char* p = ws + off;
    off = (off + bytes + 255) & ~(size_t)255;
    return p;
  };
  _Float16* feat    = (_Float16*)alloc((size_t)2048 * 1024 * 2);
  float*    embed   = (float*)   alloc((size_t)2048 * 64 * 4);
  _Float16* hz16    = (_Float16*)alloc((size_t)2048 * 160 * 2);
  float*    declin  = (float*)   alloc((size_t)2048 * 1024 * 4);
  _Float16* encPw16 = (_Float16*)alloc((size_t)64 * 1024 * 2);
  _Float16* posW1   = (_Float16*)alloc((size_t)128 * 192 * 2);
  _Float16* posW2   = (_Float16*)alloc((size_t)128 * 128 * 2);
  _Float16* posW3   = (_Float16*)alloc((size_t)64 * 128 * 2);
  _Float16* priW1   = (_Float16*)alloc((size_t)128 * 160 * 2);
  _Float16* priW2   = (_Float16*)alloc((size_t)128 * 128 * 2);
  _Float16* priW3   = (_Float16*)alloc((size_t)64 * 128 * 2);
  _Float16* gWih16  = (_Float16*)alloc((size_t)384 * 64 * 2);
  _Float16* gWhh16  = (_Float16*)alloc((size_t)384 * 128 * 2);
  _Float16* decPw16 = (_Float16*)alloc((size_t)1024 * 160 * 2);
  _Float16* encW2c  = (_Float16*)alloc((size_t)64 * 512 * 2);
  _Float16* encW3c  = (_Float16*)alloc((size_t)64 * 1024 * 2);
  _Float16* decW1p  = (_Float16*)alloc((size_t)64 * 1024 * 2);
  _Float16* decW2p  = (_Float16*)alloc((size_t)32 * 1024 * 2);

  auto cvt = [&](const float* src, _Float16* dst, int rows, int sc, int dc) {
    int n = rows * dc;
    k_cvt<<<(n + 255) / 256, 256, 0, stream>>>(src, dst, rows, sc, dc);
  };
  cvt(encPw, encPw16, 64, 1024, 1024);
  cvt(qw1, posW1, 128, 192, 192);
  cvt(qw2, posW2, 128, 128, 128);
  cvt(qw3, posW3, 64, 128, 128);
  cvt(pw1, priW1, 128, 130, 160);   // pad D+A=130 -> 160
  cvt(pw2, priW2, 128, 128, 128);
  cvt(pw3, priW3, 64, 128, 128);
  cvt(gWih, gWih16, 384, 34, 64);   // pad S+A=34 -> 64
  cvt(gWhh, gWhh16, 384, 128, 128);
  cvt(decPw, decPw16, 1024, 160, 160);
  cvt(ew2, encW2c, 64, 512, 512);   // conv weights already [oc][ic*16+j]
  cvt(ew3, encW3c, 64, 1024, 1024);
  k_cvt_dw<<<(64 * 64 * 16 + 255) / 256, 256, 0, stream>>>(dw1, decW1p, 64, 64);
  k_cvt_dw<<<(64 * 32 * 16 + 255) / 256, 256, 0, stream>>>(dw2, decW2p, 64, 32);

  float* out = (float*)d_out;
  float* o_rec  = out;                       // (B,T,1,32,32)
  float* o_zs   = out + 2097152;             // (B,T,S)
  float* o_prm  = o_zs   + 65536;
  float* o_prls = o_prm  + 65536;
  float* o_pm   = o_prls + 65536;
  float* o_pls  = o_pm   + 65536;
  float* o_hf   = o_pls  + 65536;            // (B,D)

  k_enc<<<2048, 256, 0, stream>>>(obs, ew1, eb1, encW2c, eb2, encW3c, eb3, feat);
  k_gemm_embed<<<64, 256, 0, stream>>>(feat, encPw16, encPb, embed);
  k_rssm<<<1, 256, 0, stream>>>(acts, noise, embed,
                                posW1, posW2, posW3, qb1, qb2, qb3,
                                priW1, priW2, priW3, pb1, pb2, pb3,
                                gWih16, gWhh16, gBih, gBhh,
                                hz16, o_zs, o_prm, o_prls, o_pm, o_pls, o_hf);
  k_gemm_dec<<<128, 256, 0, stream>>>(hz16, decPw16, decPb, declin);
  k_dec<<<2048, 256, 0, stream>>>(declin, decW1p, db1, decW2p, db2, dw3, db3, o_rec);
}
